// AnomalyGeneration_5781025980832
// MI455X (gfx1250) — matrix operand order repaired and checked by reading, not compiled, not run on hardware
//
#include <hip/hip_runtime.h>

typedef __attribute__((ext_vector_type(16))) __bf16 v16bf;
typedef __attribute__((ext_vector_type(8)))  __bf16 v8bf;
typedef __attribute__((ext_vector_type(8)))  float  v8f;

#define K_CODES 4096
#define EMB     128
#define KTOP    2048
#define SKIP    204
#define NCHOICES (KTOP - SKIP)   /* 1844 */
#define SC_STRIDE 4100           /* padded score row stride (dwords) to dodge bank conflicts */

__device__ __forceinline__ __bf16 f32_to_bf16(float f) {
    unsigned int u = __builtin_bit_cast(unsigned int, f);
    u += 0x7FFFu + ((u >> 16) & 1u);                 // round-to-nearest-even
    unsigned short h = (unsigned short)(u >> 16);
    return __builtin_bit_cast(__bf16, h);
}

__device__ __forceinline__ unsigned int f2ord(float f) {
    unsigned int u = __builtin_bit_cast(unsigned int, f);
    return (u & 0x80000000u) ? ~u : (u | 0x80000000u); // ascending uint == ascending float
}

__device__ __forceinline__ unsigned int hash32(unsigned int x) {
    x ^= x >> 16; x *= 0x7feb352dU;
    x ^= x >> 15; x *= 0x846ca68bU;
    x ^= x >> 16; return x;
}

// ---- codebook prep: bf16 copy + ||c||^2; one wave per code ----
__global__ void prep_cb(const float* __restrict__ cb, __bf16* __restrict__ cb_bf,
                        float* __restrict__ norms) {
    int code = blockIdx.x * 8 + (threadIdx.x >> 5);
    int lane = threadIdx.x & 31;
    float ss = 0.f;
#pragma unroll
    for (int j = 0; j < 4; ++j) {
        int d = lane + 32 * j;
        float v = cb[(size_t)code * EMB + d];
        ss += v * v;
        cb_bf[(size_t)code * EMB + d] = f32_to_bf16(v);
    }
#pragma unroll
    for (int off = 16; off > 0; off >>= 1) ss += __shfl_xor(ss, off, 32);
    if (lane == 0) norms[code] = ss;
}

// ---- mask pooling: avg_pool(>0) == any nonzero in window ----
__global__ void mask_proj(const int* __restrict__ M, float* __restrict__ mask,
                          int H, int W, int hs, int ws) {
    int idx = blockIdx.x * blockDim.x + threadIdx.x;
    if (idx >= 4 * H * W) return;
    int b = idx / (H * W), s = idx - b * (H * W);
    int h = s / W, w = s - h * W;
    int acc = 0;
    for (int i = 0; i < hs; ++i)
        for (int j = 0; j < ws; ++j)
            acc += M[((b * 128) + h * hs + i) * 256 + w * ws + j];
    mask[idx] = (acc > 0) ? 1.f : 0.f;
}

// ---- fused distance GEMM (WMMA bf16) + radix rank-select + masked scatter ----
__global__ void __launch_bounds__(256, 1)
distant_kernel(const float* __restrict__ q,
               const float* __restrict__ cb_f32,
               const __bf16* __restrict__ cb_bf,
               const float* __restrict__ norms,
               const float* __restrict__ maskp,
               float* __restrict__ outp,
               int HW, unsigned int seed) {
    extern __shared__ char smem[];
    float*        scores = (float*)smem;                                   // 16*4100 f32
    __bf16*       zl     = (__bf16*)(smem + 16 * SC_STRIDE * 4);           // 16*136 bf16
    unsigned int* hist   = (unsigned int*)(smem + 16 * SC_STRIDE * 4 + 16 * 136 * 2);
    int*          comm   = (int*)((char*)hist + 8 * 256 * 4);              // 8*64 int

    const int tid  = threadIdx.x;
    const int lane = tid & 31;
    const int wave = tid >> 5;
    const int row0 = blockIdx.x * 16;

    // stage 16x128 z tile as bf16 (gather over the strided (B,D,H,W) layout once)
    for (int t = tid; t < 16 * EMB; t += 256) {
        int m = t >> 7, d = t & 127;
        int grow = row0 + m;
        int b = grow / HW, s = grow - b * HW;
        zl[m * 136 + d] = f32_to_bf16(q[(size_t)(b * EMB + d) * HW + s]);
    }
    __syncthreads();

    // A fragments (contiguous K order, matching B): lanes 0-15 K=0..15, 16-31 K=16..31
    const int am    = lane & 15;
    const int khalf = (lane >> 4) << 4;
    v16bf afrag[4];
#pragma unroll
    for (int kt = 0; kt < 4; ++kt) {
        const __bf16* zp = zl + am * 136 + kt * 32 + khalf;
        v8bf lo = *(const v8bf*)(zp);
        v8bf hi = *(const v8bf*)(zp + 8);
        afrag[kt] = __builtin_shufflevector(lo, hi, 0,1,2,3,4,5,6,7,8,9,10,11,12,13,14,15);
    }

    // GEMM sweep: each wave covers 32 of the 256 N-tiles, two tiles per trip
    // (two independent WMMA chains hide the WMMA->VALU hazard + loop overhead).
    // score = 0.5*||c||^2 - z.c  (row norm drops out of the per-row ranking)
    const int nlane = lane & 15;
    const int klo   = (lane >> 4) << 4;   // B layout: lanes 0-15 K=0..15, 16-31 K=16..31
    const int mbase = (lane >> 4) << 3;   // C layout: VGPR v -> M = v (+8 for upper lanes)
#pragma unroll 1
    for (int i = 0; i < 16; ++i) {
        const int cb0 = (wave + 16 * i) * 16;        // N-tile pair for this trip
        const int cb1 = (wave + 8 + 16 * i) * 16;
        const __bf16* bp0 = cb_bf + (size_t)(cb0 + nlane) * EMB + klo;
        const __bf16* bp1 = cb_bf + (size_t)(cb1 + nlane) * EMB + klo;
        if (i < 15) {   // warm WGP cache for next pair (codebook is L2-resident)
            __builtin_prefetch(bp0 + (size_t)256 * EMB, 0, 3);
            __builtin_prefetch(bp1 + (size_t)256 * EMB, 0, 3);
        }
        v8f c0 = {}, c1 = {};
#pragma unroll
        for (int kt = 0; kt < 4; ++kt) {
            v16bf b0 = *(const v16bf*)(bp0 + kt * 32);
            v16bf b1 = *(const v16bf*)(bp1 + kt * 32);
            c0 = __builtin_amdgcn_wmma_f32_16x16x32_bf16(false, afrag[kt], false, b0,
                                                         (short)0, c0, false, false);
            c1 = __builtin_amdgcn_wmma_f32_16x16x32_bf16(false, afrag[kt], false, b1,
                                                         (short)0, c1, false, false);
        }
        float hn0 = 0.5f * norms[cb0 + nlane];
        float hn1 = 0.5f * norms[cb1 + nlane];
#pragma unroll
        for (int v = 0; v < 8; ++v) {
            scores[(mbase + v) * SC_STRIDE + cb0 + nlane] = hn0 - c0[v];
            scores[(mbase + v) * SC_STRIDE + cb1 + nlane] = hn1 - c1[v];
        }
    }
    __syncthreads();

    // per-row exact order-statistic via 4x8-bit radix select; wave w handles rows w, w+8
#pragma unroll 1
    for (int rr = 0; rr < 2; ++rr) {
        const int m    = wave + 8 * rr;
        const int grow = row0 + m;
        const float* srow = scores + m * SC_STRIDE;

        int r = SKIP + (int)(hash32((unsigned int)grow ^ seed) % NCHOICES);
        unsigned int prefix = 0u, vmask = 0u;
        unsigned int* h = hist + wave * 256;
        int* cm = comm + wave * 64;

#pragma unroll 1
        for (int pass = 0; pass < 4; ++pass) {
            const int shift = 24 - pass * 8;
            for (int b = lane; b < 256; b += 32) h[b] = 0u;
            __syncthreads();
#pragma unroll 1
            for (int j = 0; j < 128; ++j) {
                unsigned int u = f2ord(srow[j * 32 + lane]);
                if ((u & vmask) == prefix)
                    atomicAdd(&h[(u >> shift) & 255u], 1u);
            }
            __syncthreads();
            if (lane == 0) {
                unsigned int cum = 0; int b = 0;
                for (; b < 255; ++b) {
                    unsigned int c = h[b];
                    if (cum + c > (unsigned int)r) break;
                    cum += c;
                }
                cm[0] = b; cm[1] = r - (int)cum;
            }
            __syncthreads();
            prefix |= ((unsigned int)cm[0]) << shift;
            vmask  |= 0xFFu << shift;
            r = cm[1];
        }

        // stable tie-break: find r-th occurrence (ascending code index) of value==prefix
        const int base = lane * 128;
        int cnt = 0;
#pragma unroll 1
        for (int j = 0; j < 128; ++j)
            cnt += (f2ord(srow[base + j]) == prefix) ? 1 : 0;
        cm[8 + lane] = cnt;
        __syncthreads();
        int pre = 0;
        for (int l = 0; l < lane; ++l) pre += cm[8 + l];
        if (r >= pre && r < pre + cnt) {
            int need = r - pre;
            for (int j = 0; j < 128; ++j) {
                if (f2ord(srow[base + j]) == prefix) {
                    if (need == 0) { cm[2] = base + j; break; }
                    --need;
                }
            }
        }
        __syncthreads();
        const int sel = cm[2];

        // scatter: out = mask ? cb[sel] : q   (mask in {0,1})
        const int b = grow / HW, s = grow - b * HW;
        const bool anom = maskp[grow] > 0.f;
#pragma unroll
        for (int j = 0; j < 4; ++j) {
            int d = lane + 32 * j;
            size_t o = (size_t)(b * EMB + d) * HW + s;
            float cv = cb_f32[(size_t)sel * EMB + d];
            outp[o] = anom ? cv : q[o];
        }
        __syncthreads();
    }
}

extern "C" void kernel_launch(void* const* d_in, const int* in_sizes, int n_in,
                              void* d_out, int out_size, void* d_ws, size_t ws_size,
                              hipStream_t stream) {
    (void)in_sizes; (void)n_in; (void)out_size; (void)ws_size;
    const float* q_fine    = (const float*)d_in[0];   // (4,128,64,64)
    const float* q_coarse  = (const float*)d_in[1];   // (4,128,32,32)
    const int*   M         = (const int*)d_in[2];     // (4,1,128,256)
    const float* cb_fine   = (const float*)d_in[3];   // (4096,128)
    const float* cb_coarse = (const float*)d_in[4];   // (4096,128)

    char* ws = (char*)d_ws;
    __bf16* cbf_bf = (__bf16*)ws; ws += (size_t)K_CODES * EMB * 2;
    __bf16* cbc_bf = (__bf16*)ws; ws += (size_t)K_CODES * EMB * 2;
    float* norm_f = (float*)ws;   ws += (size_t)K_CODES * 4;
    float* norm_c = (float*)ws;   ws += (size_t)K_CODES * 4;
    float* mask_f = (float*)ws;   ws += (size_t)4 * 64 * 64 * 4;
    float* mask_c = (float*)ws;   ws += (size_t)4 * 32 * 32 * 4;

    prep_cb<<<512, 256, 0, stream>>>(cb_fine,   cbf_bf, norm_f);
    prep_cb<<<512, 256, 0, stream>>>(cb_coarse, cbc_bf, norm_c);
    mask_proj<<<64, 256, 0, stream>>>(M, mask_f, 64, 64, 2, 4);
    mask_proj<<<16, 256, 0, stream>>>(M, mask_c, 32, 32, 4, 8);

    float* out_fine   = (float*)d_out;
    float* out_coarse = out_fine + (size_t)4 * EMB * 64 * 64;

    const size_t lds = 16 * SC_STRIDE * 4 + 16 * 136 * 2 + 8 * 256 * 4 + 8 * 64 * 4; // ~277 KB
    hipFuncSetAttribute((const void*)distant_kernel,
                        hipFuncAttributeMaxDynamicSharedMemorySize, (int)lds);

    distant_kernel<<<1024, 256, lds, stream>>>(q_fine, cb_fine, cbf_bf, norm_f,
                                               mask_f, out_fine, 64 * 64, 0xA341316Cu);
    distant_kernel<<<256, 256, lds, stream>>>(q_coarse, cb_coarse, cbc_bf, norm_c,
                                              mask_c, out_coarse, 32 * 32, 0xC8013EA4u);
}
